// SelfAttention_83468394430480
// MI455X (gfx1250) — matrix operand order repaired and checked by reading, not compiled, not run on hardware
//
#include <hip/hip_runtime.h>
#include <math.h>

// ---------------------------------------------------------------------------
// Self-attention (B=4, N=64*64=4096, C=256, d=32) for gfx1250 (MI455X).
// Flash-attention streaming with v_wmma_f32_16x16x32_bf16, wave32.
// ---------------------------------------------------------------------------

typedef __bf16 v16bf __attribute__((ext_vector_type(16)));
typedef __bf16 v8bf  __attribute__((ext_vector_type(8)));
typedef float  v8f   __attribute__((ext_vector_type(8)));

#define B_   4
#define C_   256
#define N_   4096
#define D_   32
#define TOK_ (B_ * N_)   // 16384 tokens total
#define TILES_ (TOK_ / 16)  // 1024 16-token tiles

// ---- workspace layout (bytes) ----
static constexpr size_t QOFF   = 0;
static constexpr size_t KOFF   = QOFF  + (size_t)TOK_ * D_ * 2;      // +1 MiB
static constexpr size_t VTOFF  = KOFF  + (size_t)TOK_ * D_ * 2;      // +1 MiB
static constexpr size_t WQTOFF = VTOFF + (size_t)B_ * C_ * N_ * 2;   // +8 MiB
static constexpr size_t WKTOFF = WQTOFF + (size_t)D_ * C_ * 2;
static constexpr size_t WVTOFF = WKTOFF + (size_t)D_ * C_ * 2;
static constexpr size_t WSNEED = WVTOFF + (size_t)C_ * C_ * 2;       // ~10.6 MiB

// ---------------------------------------------------------------------------
// helpers
// ---------------------------------------------------------------------------
__device__ __forceinline__ v8f zero8() {
  v8f z = {0.f, 0.f, 0.f, 0.f, 0.f, 0.f, 0.f, 0.f};
  return z;
}

__device__ __forceinline__ v8f wmma_bf16(v16bf a, v16bf b, v8f c) {
  // D = A(16x32 bf16) * B(32x16 bf16) + C(16x16 f32)
  return __builtin_amdgcn_wmma_f32_16x16x32_bf16(
      /*neg_a=*/false, a, /*neg_b=*/false, b,
      /*c_mod=*/(short)0, c, /*reuse_a=*/false, /*reuse_b=*/false);
}

// max/sum reduction across the 16 lanes of each half-wave (rows of C/D tiles
// are striped across lanes 0-15 / 16-31 independently; xor masks <16 keep
// the two halves separate).
__device__ __forceinline__ float redmax16(float v) {
  v = fmaxf(v, __shfl_xor(v, 1, 32));
  v = fmaxf(v, __shfl_xor(v, 2, 32));
  v = fmaxf(v, __shfl_xor(v, 4, 32));
  v = fmaxf(v, __shfl_xor(v, 8, 32));
  return v;
}
__device__ __forceinline__ float redsum16(float v) {
  v += __shfl_xor(v, 1, 32);
  v += __shfl_xor(v, 2, 32);
  v += __shfl_xor(v, 4, 32);
  v += __shfl_xor(v, 8, 32);
  return v;
}

// Build A-operand (16x32 bf16) from row-major f32 x.
// A layout (contiguous k order, matching B): lane l holds row m=l%16;
// elems i=0..15 -> k = k0 + 16*half + i.
__device__ __forceinline__ v16bf load_a_x(const float* xrow, int k0, int half) {
  const float4* pa = (const float4*)(xrow + k0 + 16 * half);
  const float4* pb = (const float4*)(xrow + k0 + 16 * half + 8);
  float4 a0 = pa[0], a1 = pa[1];
  float4 b0 = pb[0], b1 = pb[1];
  v16bf r;
  r[0]  = (__bf16)a0.x; r[1]  = (__bf16)a0.y; r[2]  = (__bf16)a0.z; r[3]  = (__bf16)a0.w;
  r[4]  = (__bf16)a1.x; r[5]  = (__bf16)a1.y; r[6]  = (__bf16)a1.z; r[7]  = (__bf16)a1.w;
  r[8]  = (__bf16)b0.x; r[9]  = (__bf16)b0.y; r[10] = (__bf16)b0.z; r[11] = (__bf16)b0.w;
  r[12] = (__bf16)b1.x; r[13] = (__bf16)b1.y; r[14] = (__bf16)b1.z; r[15] = (__bf16)b1.w;
  return r;
}

// ---------------------------------------------------------------------------
// Kernel 0: convert + transpose weights to bf16.
//   Wq,Wk: [C][d] -> WqT,WkT: [d][C] bf16
//   Wv:    [C][C] -> WvT:     [C][C] bf16 (transposed)
// ---------------------------------------------------------------------------
__global__ __launch_bounds__(256) void wcvt_kernel(
    const float* __restrict__ Wq, const float* __restrict__ Wk,
    const float* __restrict__ Wv,
    __bf16* __restrict__ WqT, __bf16* __restrict__ WkT, __bf16* __restrict__ WvT) {
  int i = blockIdx.x * blockDim.x + threadIdx.x;
  if (i < C_ * D_) {
    int c = i / D_, n = i % D_;
    WqT[n * C_ + c] = (__bf16)Wq[i];
    WkT[n * C_ + c] = (__bf16)Wk[i];
  }
  if (i < C_ * C_) {
    int c = i / C_, n = i % C_;
    WvT[n * C_ + c] = (__bf16)Wv[i];
  }
}

// ---------------------------------------------------------------------------
// Kernel 1: projections. One wave per 16-token tile.
//   mode 0 (blockIdx.y==0): Q = x*Wq + bq, K = x*Wk + bk   (4 output tiles)
//   mode 1 (blockIdx.y==1): V = x*Wv + bv, stored transposed [B][C][N]
// ---------------------------------------------------------------------------
__global__ __launch_bounds__(128) void proj_kernel(
    const float* __restrict__ x,
    const __bf16* __restrict__ WqT, const __bf16* __restrict__ WkT,
    const __bf16* __restrict__ WvT,
    const float* __restrict__ bq, const float* __restrict__ bk,
    const float* __restrict__ bv,
    __bf16* __restrict__ Qb, __bf16* __restrict__ Kb, __bf16* __restrict__ Vt) {
  const int lane = threadIdx.x & 31;
  const int wid  = threadIdx.x >> 5;
  const int half = lane >> 4;     // 0: K-lo chunk, 1: K-hi chunk (A/B); row group (D)
  const int nIdx = lane & 15;
  const int qt   = blockIdx.x * 4 + wid;   // 0..1023
  const int tok0 = qt * 16;
  const float* xrow = x + (size_t)(tok0 + nIdx) * C_;  // A row m = nIdx

  if (blockIdx.y == 0) {
    // ---- Q and K: 4 output tiles of 16x16 ----
    v8f acc[4];
#pragma unroll
    for (int t = 0; t < 4; ++t) acc[t] = zero8();
    for (int k0 = 0; k0 < C_; k0 += 32) {
      v16bf a = load_a_x(xrow, k0, half);
#pragma unroll
      for (int t = 0; t < 4; ++t) {
        const __bf16* Wt = (t < 2) ? WqT : WkT;
        const int colbase = (t & 1) * 16;
        // B layout: lane holds col n=nIdx, elems i -> k = k0 + 16*half + i
        v16bf b = *(const v16bf*)(Wt + (size_t)(colbase + nIdx) * C_ + k0 + 16 * half);
        acc[t] = wmma_bf16(a, b, acc[t]);
      }
    }
#pragma unroll
    for (int t = 0; t < 4; ++t) {
      const float* bias = (t < 2) ? bq : bk;
      __bf16* dst = (t < 2) ? Qb : Kb;
      const int colbase = (t & 1) * 16;
      const float bb = bias[colbase + nIdx];
#pragma unroll
      for (int r = 0; r < 8; ++r) {
        // C/D layout: row m = r + 8*half, col n = nIdx
        float v = acc[t][r] + bb;
        dst[(size_t)(tok0 + r + 8 * half) * D_ + colbase + nIdx] = (__bf16)v;
      }
    }
  } else {
    // ---- V: 16 output tiles of 16x16, stored transposed [B][C][N] ----
    v8f acc[16];
#pragma unroll
    for (int t = 0; t < 16; ++t) acc[t] = zero8();
    for (int k0 = 0; k0 < C_; k0 += 32) {
      v16bf a = load_a_x(xrow, k0, half);
#pragma unroll
      for (int t = 0; t < 16; ++t) {
        v16bf b = *(const v16bf*)(WvT + (size_t)(t * 16 + nIdx) * C_ + k0 + 16 * half);
        acc[t] = wmma_bf16(a, b, acc[t]);
      }
    }
    const int bi = tok0 / N_;
    const int tl = tok0 % N_;
#pragma unroll
    for (int t = 0; t < 16; ++t) {
      const int ch = t * 16 + nIdx;
      const float bb = bv[ch];
#pragma unroll
      for (int r = 0; r < 8; ++r) {
        float v = acc[t][r] + bb;
        Vt[(size_t)(bi * C_ + ch) * N_ + tl + r + 8 * half] = (__bf16)v;
      }
    }
  }
}

// ---------------------------------------------------------------------------
// Kernel 2: flash attention. One wave per 16-query tile; stream 32 keys/iter.
//   S(16x32) = Q(16x32) * K^T  -> 2x WMMA (K-dim = d = 32)
//   online softmax (running row max / sum; rescale O accumulators)
//   O(16x256) += P(16x32) * V(32x256) -> 16x WMMA
// P crosses C/D-layout -> A-layout via wave-private LDS (per-wave DS ops are
// in-order on CDNA5, no workgroup barrier needed).
// ---------------------------------------------------------------------------
__global__ __launch_bounds__(128) void attn_kernel(
    const float* __restrict__ x,
    const __bf16* __restrict__ Qb, const __bf16* __restrict__ Kb,
    const __bf16* __restrict__ Vt,
    const float* __restrict__ gamma, float* __restrict__ out) {
  __shared__ __align__(64) __bf16 lds_p[4][16][32];   // 4 waves x 1 KB

  const int lane = threadIdx.x & 31;
  const int wid  = threadIdx.x >> 5;
  const int half = lane >> 4;
  const int nIdx = lane & 15;
  const int qt   = blockIdx.x * 4 + wid;   // 0..1023
  const int bi   = qt >> 8;                // batch = qt*16 / 4096

  // A-operand: Q rows (bf16, row-major [tok][32])
  const __bf16* qrow = Qb + (size_t)(qt * 16 + nIdx) * D_;
  v8bf qa0 = *(const v8bf*)(qrow + 16 * half);
  v8bf qa1 = *(const v8bf*)(qrow + 16 * half + 8);
  v16bf aq;
#pragma unroll
  for (int i = 0; i < 8; ++i) { aq[i] = qa0[i]; aq[i + 8] = qa1[i]; }

  float rowmax[8], rowsum[8];
#pragma unroll
  for (int r = 0; r < 8; ++r) { rowmax[r] = -__builtin_inff(); rowsum[r] = 0.f; }
  v8f acc[16];
#pragma unroll
  for (int t = 0; t < 16; ++t) acc[t] = zero8();

  const __bf16* Kbase = Kb + (size_t)bi * N_ * D_;
  const __bf16* Vbase = Vt + (size_t)bi * C_ * N_;

  for (int j = 0; j < N_; j += 32) {
    // B-operand = K^T: lane holds key n, elems i -> feature 16*half + i
    v16bf b0 = *(const v16bf*)(Kbase + (size_t)(j + nIdx) * D_ + 16 * half);
    v16bf b1 = *(const v16bf*)(Kbase + (size_t)(j + 16 + nIdx) * D_ + 16 * half);
    v8f s0 = wmma_bf16(aq, b0, zero8());   // scores, keys j..j+15
    v8f s1 = wmma_bf16(aq, b1, zero8());   // scores, keys j+16..j+31

    float sc[8], e0[8], e1[8];
#pragma unroll
    for (int r = 0; r < 8; ++r) {
      float mx = redmax16(fmaxf(s0[r], s1[r]));
      float nm = fmaxf(rowmax[r], mx);
      sc[r] = __expf(rowmax[r] - nm);
      e0[r] = __expf(s0[r] - nm);
      e1[r] = __expf(s1[r] - nm);
      float bs = redsum16(e0[r] + e1[r]);
      rowsum[r] = rowsum[r] * sc[r] + bs;
      rowmax[r] = nm;
    }
#pragma unroll
    for (int t = 0; t < 16; ++t)
#pragma unroll
      for (int r = 0; r < 8; ++r) acc[t][r] *= sc[r];

    // Stage P (C/D layout) to LDS, reload in A layout (wave-private region).
#pragma unroll
    for (int r = 0; r < 8; ++r) {
      const int m = r + 8 * half;
      lds_p[wid][m][nIdx]      = (__bf16)e0[r];
      lds_p[wid][m][16 + nIdx] = (__bf16)e1[r];
    }
    v8bf pa0 = *(const v8bf*)&lds_p[wid][nIdx][16 * half];
    v8bf pa1 = *(const v8bf*)&lds_p[wid][nIdx][16 * half + 8];
    v16bf ap;
#pragma unroll
    for (int i = 0; i < 8; ++i) { ap[i] = pa0[i]; ap[i + 8] = pa1[i]; }

    if (j + 32 < N_) {  // uniform branch (j is wave-uniform)
      __builtin_prefetch(Kbase + (size_t)(j + 32 + nIdx) * D_, 0, 1);
    }

    // O += P * V  (V transposed: lane holds channel, elems i -> key offset)
#pragma unroll
    for (int t = 0; t < 16; ++t) {
      v16bf bv = *(const v16bf*)(Vbase + (size_t)(t * 16 + nIdx) * N_ + j + 16 * half);
      acc[t] = wmma_bf16(ap, bv, acc[t]);
    }
  }

  const float g = gamma[0];
#pragma unroll
  for (int t = 0; t < 16; ++t) {
#pragma unroll
    for (int r = 0; r < 8; ++r) {
      const size_t tok = (size_t)qt * 16 + r + 8 * half;
      const size_t idx = tok * C_ + t * 16 + nIdx;
      out[idx] = g * (acc[t][r] / rowsum[r]) + x[idx];
    }
  }
}

// ---------------------------------------------------------------------------
// Host launch
// ---------------------------------------------------------------------------
extern "C" void kernel_launch(void* const* d_in, const int* in_sizes, int n_in,
                              void* d_out, int out_size, void* d_ws, size_t ws_size,
                              hipStream_t stream) {
  const float* x     = (const float*)d_in[0];
  const float* Wq    = (const float*)d_in[1];
  const float* bq    = (const float*)d_in[2];
  const float* Wk    = (const float*)d_in[3];
  const float* bk    = (const float*)d_in[4];
  const float* Wv    = (const float*)d_in[5];
  const float* bv    = (const float*)d_in[6];
  const float* gamma = (const float*)d_in[7];
  float* out = (float*)d_out;

  char* ws = (char*)d_ws;
  __bf16* Qb  = (__bf16*)(ws + QOFF);
  __bf16* Kb  = (__bf16*)(ws + KOFF);
  __bf16* Vt  = (__bf16*)(ws + VTOFF);
  __bf16* WqT = (__bf16*)(ws + WQTOFF);
  __bf16* WkT = (__bf16*)(ws + WKTOFF);
  __bf16* WvT = (__bf16*)(ws + WVTOFF);
  (void)in_sizes; (void)n_in; (void)out_size; (void)ws_size; // needs ~10.6 MiB ws

  // 0) weights -> bf16, transposed
  wcvt_kernel<<<(C_ * C_ + 255) / 256, 256, 0, stream>>>(Wq, Wk, Wv, WqT, WkT, WvT);
  // 1) projections (grid.y: 0 = Q/K, 1 = V)
  proj_kernel<<<dim3(TILES_ / 4, 2), 128, 0, stream>>>(x, WqT, WkT, WvT, bq, bk, bv,
                                                       Qb, Kb, Vt);
  // 2) flash attention + gamma*out + x
  attn_kernel<<<TILES_ / 4, 128, 0, stream>>>(x, Qb, Kb, Vt, gamma, out);
}
